// BoundaryActivation_18408229831218
// MI455X (gfx1250) — matrix operand order repaired and checked by reading, not compiled, not run on hardware
//
#include <hip/hip_runtime.h>

typedef __bf16 bf16_t;
typedef bf16_t bfv16 __attribute__((ext_vector_type(16)));
typedef bf16_t bfv8  __attribute__((ext_vector_type(8)));
typedef float  fv8   __attribute__((ext_vector_type(8)));
typedef float  fv4   __attribute__((ext_vector_type(4)));
typedef unsigned int v4u __attribute__((ext_vector_type(4)));
typedef int          v8i __attribute__((ext_vector_type(8)));
typedef int          v4i __attribute__((ext_vector_type(4)));

#define C_IN  256
#define KDIM  1280          // 5*C
#define HW    16384         // 128*128
#define NSTEP 40            // K steps of 32
#define LROW  272           // padded LDS bytes per B-tile row (32 rows)
#define TILEB (32 * LROW)   // bytes per LDS B buffer

// ---------------------------------------------------------------------------
// Kernel 1: weight swizzle  f32 [256][1280] -> bf16 A-fragment order
// wsA layout: [otile(16)][ktile(40)][lane(32)][elem(16)]
// ISA 16-bit A 16x32 layout: lanes 0-15 M=lane, group=lane/16 adds K+8;
// VGPR v (elem pair e/2): K = (v<4 ? 2v : 16+2(v-4)) + 8*group + (e&1)
// ---------------------------------------------------------------------------
__global__ __launch_bounds__(256) void swz_w(const float* __restrict__ w,
                                             bf16_t* __restrict__ wsA) {
  int g    = blockIdx.x * 256 + threadIdx.x;     // 327680 total
  int e    =  g        & 15;
  int lane = (g >> 4)  & 31;
  int kt   = (g >> 9)  % 40;
  int ot   =  g / (512 * 40);
  int grp  = lane >> 4;
  int v    = e >> 1;
  int kk   = (v < 4 ? 2 * v : 16 + 2 * (v - 4)) + 8 * grp + (e & 1);
  int o    = ot * 16 + (lane & 15);
  int k    = kt * 32 + kk;
  wsA[g] = (bf16_t)w[o * KDIM + k];
}

// ---------------------------------------------------------------------------
// Kernel 2: column scans along H: f=0 copy, f=1 cummax down, f=2 suffix up.
// One thread per (b,c,w); coalesced over W.
// ---------------------------------------------------------------------------
__global__ __launch_bounds__(256) void scan_h(const float* __restrict__ x,
                                              bf16_t* __restrict__ wsX) {
  int g = blockIdx.x * 256 + threadIdx.x;        // 262144 total
  int w = g & 127;
  int c = (g >> 7) & 255;
  int b = g >> 15;
  size_t xbase = ((size_t)(b * C_IN + c)) * HW + w;
  size_t f0 = ((size_t)(b * 5 + 0) * C_IN + c) * HW + w;
  size_t f1 = f0 + (size_t)C_IN * HW;
  size_t f2 = f1 + (size_t)C_IN * HW;
  float m = -3.402823466e38f;
  for (int h = 0; h < 128; ++h) {
    float v = x[xbase + (size_t)h * 128];
    m = fmaxf(m, v);
    wsX[f0 + (size_t)h * 128] = (bf16_t)v;
    wsX[f1 + (size_t)h * 128] = (bf16_t)m;
  }
  m = -3.402823466e38f;
  for (int h = 127; h >= 0; --h) {
    float v = x[xbase + (size_t)h * 128];
    m = fmaxf(m, v);
    wsX[f2 + (size_t)h * 128] = (bf16_t)m;
  }
}

// ---------------------------------------------------------------------------
// Kernel 3: row scans along W via LDS tile (64 rows of one (b,c) plane).
// f=3 cummax L->R, f=4 suffix R->L. 128 threads/WG.
// ---------------------------------------------------------------------------
__global__ __launch_bounds__(128) void scan_w(const float* __restrict__ x,
                                              bf16_t* __restrict__ wsX) {
  __shared__ bf16_t plane[64 * 132];
  __shared__ bf16_t o3[64 * 130];
  __shared__ bf16_t o4[64 * 130];
  int t    = threadIdx.x;              // 0..127
  int blk  = blockIdx.x;               // 4096
  int half = blk & 1;
  int bc   = blk >> 1;
  int c = bc & 255, b = bc >> 8;
  size_t gbase = ((size_t)(b * C_IN + c)) * HW + (size_t)half * 64 * 128;

  for (int i = 0; i < 16; ++i) {
    int idx = i * 512 + t * 4;
    fv4 v = *(const fv4*)(x + gbase + idx);
    int r = idx >> 7, col = idx & 127;
#pragma unroll
    for (int q = 0; q < 4; ++q) plane[r * 132 + col + q] = (bf16_t)v[q];
  }
  __syncthreads();

  if (t < 64) {
    int r = t;
    float m = -3.402823466e38f;
    for (int w = 0; w < 128; ++w) {
      m = fmaxf(m, (float)plane[r * 132 + w]);
      o3[r * 130 + w] = (bf16_t)m;
    }
  } else {
    int r = t - 64;
    float m = -3.402823466e38f;
    for (int w = 127; w >= 0; --w) {
      m = fmaxf(m, (float)plane[r * 132 + w]);
      o4[r * 130 + w] = (bf16_t)m;
    }
  }
  __syncthreads();

  size_t f3 = ((size_t)(b * 5 + 3) * C_IN + c) * HW + (size_t)half * 64 * 128;
  size_t f4 = f3 + (size_t)C_IN * HW;
  int row = t >> 1, colb = (t & 1) * 64;
  for (int i = 0; i < 64; ++i) {
    wsX[f3 + (size_t)row * 128 + colb + i] = o3[row * 130 + colb + i];
    wsX[f4 + (size_t)row * 128 + colb + i] = o4[row * 130 + colb + i];
  }
}

// ---------------------------------------------------------------------------
// TDM: DMA one 32(K) x 64(N) bf16 tile global->LDS with hardware padding so
// each 128B row lands on a 272B-stride LDS row (bank-friendly fragments).
// D# per CDNA5 ISA:
//  group0: [1:0]=count=1, [63:32]=lds_addr, [120:64]=global_addr, [127:126]=2
//  group1: data_size=1(2B), pad_enable, pad_interval=4 (32 dw = 128B),
//          pad_amount=35 (36 dw = 144B), tensor_dim0=16384, tensor_dim1=1M,
//          tile_dim0=64, tile_dim1=32, tensor_dim0_stride=16384
// Toolchain is the 6-arg builtin lane: (v4u, v8i, v4i, v4i, v8i, i32 cpol).
// ---------------------------------------------------------------------------
__device__ __forceinline__ void tdm_load_tile(unsigned lds_byte_addr,
                                              const bf16_t* gptr) {
  unsigned long long ga = (unsigned long long)(uintptr_t)gptr;
  v4u g0;
  g0[0] = 1u;                                            // count=1
  g0[1] = lds_byte_addr;                                 // lds_addr
  g0[2] = (unsigned)ga;                                  // global_addr lo
  g0[3] = ((unsigned)(ga >> 32) & 0x01FFFFFFu) | (2u << 30);  // hi | type=2
  v8i g1;
  g1[0] = (int)((1u << 16) | (1u << 20) | (4u << 22) | (35u << 25));
  g1[1] = (int)(16384u << 16);                           // tensor_dim0 lo16
  g1[2] = 0;                                             // dim0 hi / dim1 lo
  g1[3] = (int)(0x10u | (64u << 16));                    // dim1 hi | tile_dim0
  g1[4] = 32;                                            // tile_dim1 (tile_dim2=0)
  g1[5] = 16384;                                         // dim0_stride lo
  g1[6] = 0;
  g1[7] = 0;
  v4i z4 = {0, 0, 0, 0};
  v8i z8 = {0, 0, 0, 0, 0, 0, 0, 0};
  __builtin_amdgcn_tensor_load_to_lds(g0, g1, z4, z4, z8, 0);
}

// ---------------------------------------------------------------------------
// Kernel 4: GEMM  out[b,o,n] = sum_k W[o,k]*X[b,k,n] + bias[o]
// M=256 (16 waves x 16), N=64/WG, K=1280 in 40 steps of 32.
// B tiles double-buffered in LDS via TDM; v_wmma_f32_16x16x32_bf16.
// ---------------------------------------------------------------------------
__global__ __launch_bounds__(512) void gemm(const bf16_t* __restrict__ wsA,
                                            const bf16_t* __restrict__ wsX,
                                            const float* __restrict__ bias,
                                            float* __restrict__ out) {
  __shared__ __align__(16) char ldsB[2][TILEB];
  int t    = threadIdx.x;
  int wave = t >> 5, lane = t & 31;
  int blk  = blockIdx.x;
  int b    = blk >> 8;
  int n0   = (blk & 255) * 64;

  const bf16_t* Xb   = wsX + (size_t)b * KDIM * HW + n0;   // row 0 of K-tile 0
  unsigned lds0 = (unsigned)(unsigned long long)
      (__attribute__((address_space(3))) char*)&ldsB[0][0];

  fv8 acc0 = {}, acc1 = {}, acc2 = {}, acc3 = {};
  const bfv16* Aall = (const bfv16*)wsA;

  // prologue: DMA tile 0 into buffer 0
  if (wave == 0) {
    tdm_load_tile(lds0, Xb);
    __builtin_amdgcn_s_wait_tensorcnt(0);
  }
  __syncthreads();

  for (int kt = 0; kt < NSTEP; ++kt) {
    int cur = kt & 1;
    if (wave == 0 && kt + 1 < NSTEP)
      tdm_load_tile(lds0 + (unsigned)(cur ^ 1) * TILEB,
                    Xb + (size_t)(kt + 1) * 32 * HW);

    bfv16 a = Aall[(wave * NSTEP + kt) * 32 + lane];
    if (kt + 1 < NSTEP)
      __builtin_prefetch(&Aall[(wave * NSTEP + kt + 1) * 32 + lane], 0, 3);

    // hoist all 4 B fragments: one ds clause, one wait, then 4 WMMAs
    const char* basep = &ldsB[cur][(lane & 15) * 2];
    int kb = (lane >> 4) * 8;
    bfv16 b0, b1, b2, b3;
#pragma unroll
    for (int s = 0; s < 16; ++s) {
      const char* p = basep + (kb + (s & 7) + (s >> 3) * 16) * LROW;
      b0[s] = *(const bf16_t*)(p +  0);
      b1[s] = *(const bf16_t*)(p + 32);
      b2[s] = *(const bf16_t*)(p + 64);
      b3[s] = *(const bf16_t*)(p + 96);
    }
    acc0 = __builtin_amdgcn_wmma_f32_16x16x32_bf16(false, a, false, b0,
                                                   (short)0, acc0, false, false);
    acc1 = __builtin_amdgcn_wmma_f32_16x16x32_bf16(false, a, false, b1,
                                                   (short)0, acc1, false, false);
    acc2 = __builtin_amdgcn_wmma_f32_16x16x32_bf16(false, a, false, b2,
                                                   (short)0, acc2, false, false);
    acc3 = __builtin_amdgcn_wmma_f32_16x16x32_bf16(false, a, false, b3,
                                                   (short)0, acc3, false, false);

    if (wave == 0)
      __builtin_amdgcn_s_wait_tensorcnt(0);
    __syncthreads();
  }

  // epilogue: C layout  M = r + 8*(lane/16), N = lane%16
  int grp = lane >> 4;
  int nn  = lane & 15;
  float bv[8];
#pragma unroll
  for (int r = 0; r < 8; ++r) bv[r] = bias[wave * 16 + r + 8 * grp];
  float* ob = out + (size_t)b * C_IN * HW;
#pragma unroll
  for (int j = 0; j < 4; ++j) {
    fv8 acc = (j == 0) ? acc0 : (j == 1) ? acc1 : (j == 2) ? acc2 : acc3;
#pragma unroll
    for (int r = 0; r < 8; ++r) {
      int o = wave * 16 + r + 8 * grp;
      ob[(size_t)o * HW + n0 + j * 16 + nn] = acc[r] + bv[r];
    }
  }
}

// ---------------------------------------------------------------------------
extern "C" void kernel_launch(void* const* d_in, const int* in_sizes, int n_in,
                              void* d_out, int out_size, void* d_ws, size_t ws_size,
                              hipStream_t stream) {
  const float* x  = (const float*)d_in[0];   // [8,256,128,128]
  const float* cw = (const float*)d_in[1];   // [256,1280]
  const float* cb = (const float*)d_in[2];   // [256]
  float* out = (float*)d_out;                // [8,256,128,128]

  char*   wsb = (char*)d_ws;
  bf16_t* wsA = (bf16_t*)wsb;                      // 655 KB swizzled weights
  bf16_t* wsX = (bf16_t*)(wsb + (1 << 20));        // 335 MB staged features

  swz_w <<<1280, 256, 0, stream>>>(cw, wsA);
  scan_h<<<1024, 256, 0, stream>>>(x, wsX);
  scan_w<<<4096, 128, 0, stream>>>(x, wsX);
  gemm  <<<2048, 512, 0, stream>>>(wsA, wsX, cb, out);
}
